// E3nnProteinModel_19722489823976
// MI455X (gfx1250) — matrix operand order repaired and checked by reading, not compiled, not run on hardware
//
#include <hip/hip_runtime.h>
#include <hip/hip_bf16.h>
#include <stdint.h>

// ---------------------------------------------------------------------------
// E3nn protein GNN forward, algebraically reordered:
//   conv(h) = (scatter_mean(h[src], dst)) @ W * TP_NORM     (matmul is linear)
// GEMMs: v_wmma_f32_16x16x32_bf16, A staged fp32->bf16 through LDS (shared by
// all 8 waves of the block), B pre-transposed+pre-converted to bf16, per-row
// mean scaling folded into the fp32 epilogue.
// ---------------------------------------------------------------------------

typedef __attribute__((ext_vector_type(16))) __bf16 v16bf;
typedef __attribute__((ext_vector_type(8)))  float  v8f;

#define NN     50000
#define EE     1600000
#define F_IN   1280
#define HD     128
#define GG     64
#define OUTD   2
#define MREP   4                 // row-tiles per wave -> 64 rows per block
#define ROWDW  20                // LDS row stride in dwords (80B: pad vs bank conflicts)
#define TP_NORM 0.0249338895f    // (0.5/sqrt(pi)) / sqrt(128)

// ---------------------------------------------------------------------------
// GEMM: out[m,n] = act( outScale * inv[m] * sum_k A[m,k]*Wt[n,k] + bias[n] )
//   inv[m] = 1/max(rowCnt[m],1) if rowCnt else 1   (applied in fp32 epilogue)
// Wt: bf16, pre-transposed [128][K]. Block = 256 thr = 8 waves; wave w owns
// columns [16w,16w+16); MREP stacked 16x16 row-tiles share one B fragment.
// K multiple of 32; ncols fixed 128.
// ---------------------------------------------------------------------------
__global__ __launch_bounds__(256) void wmma_gemm_bf16(
    const float* __restrict__ A, const __bf16* __restrict__ Wt,
    const float* __restrict__ bias, const float* __restrict__ rowCnt,
    float* __restrict__ out, int Nrows, int K, float outScale, int doRelu)
{
  __shared__ uint32_t ldsA[64 * ROWDW];     // 64 rows x 32 bf16 (padded to 80B)

  const int lane = threadIdx.x & 31;
  const int wave = threadIdx.x >> 5;
  const int half = lane >> 4;               // K-half selector (A/B frag layout)
  const int r    = lane & 15;
  const int n    = wave * 16 + r;           // output column
  const long blockRow = (long)blockIdx.x * (16 * MREP);

  // staging role: thread t loads row (t>>2), k-quad (t&3) of the A tile
  const int srow = threadIdx.x >> 2;
  const int kq   = threadIdx.x & 3;
  long mstage = blockRow + srow;
  if (mstage >= Nrows) mstage = Nrows - 1;  // clamp; stores guarded later
  const float* arow = A + (size_t)mstage * K;

  v8f acc[MREP] = {};

  const int ksteps = K >> 5;
  for (int kb = 0; kb < ksteps; ++kb) {
    const int k0 = kb << 5;

    // ---- cooperative stage: global fp32 -> LDS bf16 (tile read ONCE/block) --
    __syncthreads();                                   // frag reads of prev step done
    {
      const float4* ap = (const float4*)(arow + k0 + (kq << 3));
      float4 a0 = ap[0], a1 = ap[1];
      union { uint4 u; __bf16 h[8]; } pk;
      pk.h[0]=(__bf16)a0.x; pk.h[1]=(__bf16)a0.y; pk.h[2]=(__bf16)a0.z; pk.h[3]=(__bf16)a0.w;
      pk.h[4]=(__bf16)a1.x; pk.h[5]=(__bf16)a1.y; pk.h[6]=(__bf16)a1.z; pk.h[7]=(__bf16)a1.w;
      *(uint4*)(ldsA + srow * ROWDW + (kq << 2)) = pk.u;
    }
    __syncthreads();                                   // tile staged

    // ---- B fragment: col n, k = k0 + half*16 + j (16 contiguous bf16) ------
    const v16bf bfrag = *(const v16bf*)(Wt + (size_t)n * K + k0 + (half << 4));

#pragma unroll
    for (int f = 0; f < MREP; ++f) {
      // A fragment from LDS: row f*16+r; k = half*16 + j (16 contiguous bf16)
      const uint32_t* rp = ldsA + (f * 16 + r) * ROWDW;
      union { v16bf v; uint4 u[2]; } af;
      af.u[0] = *(const uint4*)(rp + (half << 3));         // bytes half*32
      af.u[1] = *(const uint4*)(rp + 4 + (half << 3));     // bytes 16+half*32
      acc[f] = __builtin_amdgcn_wmma_f32_16x16x32_bf16(
          /*neg_a=*/false, af.v, /*neg_b=*/false, bfrag,
          /*c_mod=*/(short)0, acc[f], /*reuse_a=*/false, /*reuse_b=*/false);
    }
  }

  // ---- epilogue: C layout -> VGPR rr = row (half*8+rr), col (lane&15) -------
  const float b = bias ? bias[n] : 0.0f;
#pragma unroll
  for (int f = 0; f < MREP; ++f) {
#pragma unroll
    for (int rr = 0; rr < 8; ++rr) {
      long m = blockRow + f * 16 + (half << 3) + rr;
      if (m < Nrows) {
        float v = acc[f][rr];
        if (rowCnt) v *= 1.0f / fmaxf(rowCnt[m], 1.0f);   // scatter-mean divide
        v = v * outScale + b;
        if (doRelu) v = fmaxf(v, 0.0f);
        out[m * HD + n] = v;
      }
    }
  }
}

// ---------------------------------------------------------------------------
// Weight transpose + fp32->bf16: Wt[n*K + k] = bf16(W[k*128 + n])
// ---------------------------------------------------------------------------
__global__ void transpose_w(const float* __restrict__ W, __bf16* __restrict__ Wt, int K)
{
  int idx = blockIdx.x * blockDim.x + threadIdx.x;
  if (idx >= K * HD) return;
  int k = idx / HD, n = idx % HD;
  Wt[(size_t)n * K + k] = (__bf16)W[idx];
}

// ---------------------------------------------------------------------------
// Edge scatter-add: s[dst] += h[src] (float4 per thread), cnt[dst] += 1
// ---------------------------------------------------------------------------
__global__ void scatter_edges(const float* __restrict__ h, const int* __restrict__ src,
                              const int* __restrict__ dst, float* __restrict__ s,
                              float* __restrict__ cnt, long total)
{
  long idx = (long)blockIdx.x * blockDim.x + threadIdx.x;
  if (idx >= total) return;
  int e  = (int)(idx >> 5);
  int c4 = (int)(idx & 31) << 2;
  int sn = src[e], dn = dst[e];
  const float4 v = *(const float4*)(h + (size_t)sn * HD + c4);
  float* p = s + (size_t)dn * HD + c4;
  atomicAdd(p + 0, v.x); atomicAdd(p + 1, v.y);
  atomicAdd(p + 2, v.z); atomicAdd(p + 3, v.w);
  if ((idx & 31) == 0) atomicAdd(cnt + dn, 1.0f);
}

// ---------------------------------------------------------------------------
// Graph pooling: zs[batch[i]] += h[i], cntg[batch[i]] += 1
// ---------------------------------------------------------------------------
__global__ void pool_nodes(const float* __restrict__ h, const int* __restrict__ batch,
                           float* __restrict__ zs, float* __restrict__ cntg, long total)
{
  long idx = (long)blockIdx.x * blockDim.x + threadIdx.x;
  if (idx >= total) return;
  int i  = (int)(idx >> 5);
  int c4 = (int)(idx & 31) << 2;
  int g  = batch[i];
  const float4 v = *(const float4*)(h + (size_t)i * HD + c4);
  float* p = zs + (size_t)g * HD + c4;
  atomicAdd(p + 0, v.x); atomicAdd(p + 1, v.y);
  atomicAdd(p + 2, v.z); atomicAdd(p + 3, v.w);
  if ((idx & 31) == 0) atomicAdd(cntg + g, 1.0f);
}

__global__ void finalize_z(const float* __restrict__ zs, const float* __restrict__ cntg,
                           float* __restrict__ zout)
{
  int idx = blockIdx.x * blockDim.x + threadIdx.x;
  if (idx >= GG * HD) return;
  zout[idx] = zs[idx] / fmaxf(cntg[idx >> 7], 1.0f);
}

// Final tiny layer: out[g,o] = o2[g,:] @ c3w[:,o] + c3b[o]   (64x2)
__global__ void classifier_out(const float* __restrict__ o2, const float* __restrict__ c3w,
                               const float* __restrict__ c3b, float* __restrict__ outp)
{
  int idx = threadIdx.x;          // 128 threads
  int g = idx >> 1, o = idx & 1;
  float s = c3b[o];
  for (int k = 0; k < HD; ++k) s += o2[g * HD + k] * c3w[k * OUTD + o];
  outp[idx] = s;
}

extern "C" void kernel_launch(void* const* d_in, const int* in_sizes, int n_in,
                              void* d_out, int out_size, void* d_ws, size_t ws_size,
                              hipStream_t stream) {
  const float* x     = (const float*)d_in[0];
  const int*   eidx  = (const int*)  d_in[2];
  const int*   batch = (const int*)  d_in[3];
  const float* lin_w = (const float*)d_in[4];
  const float* lin_b = (const float*)d_in[5];
  const float* tpw[3]= {(const float*)d_in[6], (const float*)d_in[7], (const float*)d_in[8]};
  const float* c1w   = (const float*)d_in[9];
  const float* c1b   = (const float*)d_in[10];
  const float* c2w   = (const float*)d_in[11];
  const float* c2b   = (const float*)d_in[12];
  const float* c3w   = (const float*)d_in[13];
  const float* c3b   = (const float*)d_in[14];

  const int* src = eidx;
  const int* dst = eidx + EE;

  // ---- workspace layout, all regions 16B aligned ----
  float* X     = (float*)d_ws;                 // [N,128] current features
  float* Y     = X   + (size_t)NN * HD;        // [N,128] scatter accumulator
  float* cnt   = Y   + (size_t)NN * HD;        // [N]
  float* zs    = cnt + NN;                     // [G,128]
  float* cntg  = zs  + GG * HD;                // [G]
  float* o1    = cntg+ GG;                     // [G,128]
  float* o2    = o1  + GG * HD;                // [G,128]
  __bf16* WtLin = (__bf16*)(o2 + GG * HD);     // [128,1280] bf16
  __bf16* Wt1   = WtLin + (size_t)HD * F_IN;   // [128,128] bf16 x5
  __bf16* Wt2   = Wt1 + HD * HD;
  __bf16* Wt3   = Wt2 + HD * HD;
  __bf16* WtC1  = Wt3 + HD * HD;
  __bf16* WtC2  = WtC1 + HD * HD;

  float* dout = (float*)d_out;
  float* zout = dout + GG * OUTD;              // z occupies d_out[128 .. 128+8192)

  // ---- pre-transpose + bf16-convert weights ----
  transpose_w<<<(F_IN * HD + 255) / 256, 256, 0, stream>>>(lin_w, WtLin, F_IN);
  transpose_w<<<(HD * HD + 255) / 256, 256, 0, stream>>>(tpw[0], Wt1, HD);
  transpose_w<<<(HD * HD + 255) / 256, 256, 0, stream>>>(tpw[1], Wt2, HD);
  transpose_w<<<(HD * HD + 255) / 256, 256, 0, stream>>>(tpw[2], Wt3, HD);
  transpose_w<<<(HD * HD + 255) / 256, 256, 0, stream>>>(c1w, WtC1, HD);
  transpose_w<<<(HD * HD + 255) / 256, 256, 0, stream>>>(c2w, WtC2, HD);

  const int rowsPerBlk = 16 * MREP;
  const int gemmBlocksN = (NN + rowsPerBlk - 1) / rowsPerBlk;

  // ---- lin_in + relu:  X = relu(x @ lin_w + lin_b) ----
  wmma_gemm_bf16<<<gemmBlocksN, 256, 0, stream>>>(
      x, WtLin, lin_b, nullptr, X, NN, F_IN, 1.0f, 1);

  // ---- 3 conv layers: scatter-mean then GEMM (mean divide in epilogue) ----
  const long scatTotal = (long)EE * 32;
  const int  scatBlocks = (int)((scatTotal + 255) / 256);
  __bf16* WtTp[3] = {Wt1, Wt2, Wt3};
  for (int l = 0; l < 3; ++l) {
    hipMemsetAsync(Y,   0, (size_t)NN * HD * sizeof(float), stream);
    hipMemsetAsync(cnt, 0, (size_t)NN * sizeof(float),      stream);
    scatter_edges<<<scatBlocks, 256, 0, stream>>>(X, src, dst, Y, cnt, scatTotal);
    wmma_gemm_bf16<<<gemmBlocksN, 256, 0, stream>>>(
        Y, WtTp[l], nullptr, cnt, X, NN, HD, TP_NORM, (l < 2) ? 1 : 0);
  }

  // ---- global mean pool -> z (written straight into d_out tail) ----
  hipMemsetAsync(zs,   0, (size_t)GG * HD * sizeof(float), stream);
  hipMemsetAsync(cntg, 0, (size_t)GG * sizeof(float),      stream);
  const long poolTotal = (long)NN * 32;
  pool_nodes<<<(int)((poolTotal + 255) / 256), 256, 0, stream>>>(X, batch, zs, cntg, poolTotal);
  finalize_z<<<(GG * HD + 255) / 256, 256, 0, stream>>>(zs, cntg, zout);

  // ---- classifier MLP ----
  wmma_gemm_bf16<<<1, 256, 0, stream>>>(zout, WtC1, c1b, nullptr, o1, GG, HD, 1.0f, 1);
  wmma_gemm_bf16<<<1, 256, 0, stream>>>(o1,   WtC2, c2b, nullptr, o2, GG, HD, 1.0f, 1);
  classifier_out<<<1, 128, 0, stream>>>(o2, c3w, c3b, dout);
}
